// ScRRAMBLeCIFAR_36567351558788
// MI455X (gfx1250) — matrix operand order repaired and checked by reading, not compiled, not run on hardware
//
#include <hip/hip_runtime.h>
#include <hip/hip_bf16.h>

// ---------------------------------------------------------------------------
// Types for CDNA5 WMMA (wave32): V_WMMA_F32_16X16X32_BF16
// ---------------------------------------------------------------------------
typedef __attribute__((ext_vector_type(16))) __bf16 bf16x16;
typedef __attribute__((ext_vector_type(8)))  float  f32x8;

#define BATCH   512
#define INFLAT  3072
#define PARFLAT 2560

// ---------------------------------------------------------------------------
// Capsule layer: out[b,k,l,n] = relu( sum_m (sum_p C[p,k,l]*x[b,p*64+m]) * W[k,l,m,n] + bias[k,l,n] )
// Optionally fused skip term: + sum_p Cskip[p,k,l]*x[b,p*64+n]
// One 64-thread block per (b, k*4+l). Input row staged once in LDS.
// Exact fp32 — this layer is <2% of total FLOPs, not worth WMMA plumbing.
// ---------------------------------------------------------------------------
template<int KOUT, bool DO_SKIP>
__global__ void caps_layer(const float* __restrict__ x,
                           const float* __restrict__ C,
                           const float* __restrict__ W,
                           const float* __restrict__ bias,
                           const float* __restrict__ Cskip,
                           float* __restrict__ out) {
  __shared__ float xs[INFLAT];
  __shared__ float routed[64];

  const int b  = blockIdx.x;
  const int kl = blockIdx.y;        // 0 .. KOUT*4-1
  const int kc = kl >> 2;
  const int l  = kl & 3;
  const int t  = threadIdx.x;       // 0..63

  for (int i = t; i < INFLAT; i += 64) xs[i] = x[(size_t)b * INFLAT + i];
  __syncthreads();

  // routing: routed[m] = sum_{p=0..47} C[i,j,kc,l] * x[b, p*64+m]
  float r = 0.f;
  #pragma unroll
  for (int p = 0; p < 48; ++p) {
    const float cv = C[(p * KOUT + kc) * 4 + l];   // uniform -> scalar loads
    r = fmaf(cv, xs[p * 64 + t], r);
  }
  routed[t] = r;
  __syncthreads();

  // 64x64 weight apply + bias + relu
  float s = bias[kl * 64 + t];
  #pragma unroll 8
  for (int m = 0; m < 64; ++m)
    s = fmaf(routed[m], W[((size_t)kl * 64 + m) * 64 + t], s);
  s = fmaxf(s, 0.f);

  if (DO_SKIP) {
    float sk = 0.f;
    #pragma unroll
    for (int p = 0; p < 48; ++p)
      sk = fmaf(Cskip[(p * KOUT + kc) * 4 + l], xs[p * 64 + t], sk);
    s += sk;
  }
  out[(size_t)b * (KOUT * 256) + kl * 64 + t] = s;
}

// ---------------------------------------------------------------------------
// Top-1 capsule mask: per batch row, argmax of capsule squared-norms (10 caps
// of 256), masked copy x_sum -> x_recon. One 256-thread block per row.
// ---------------------------------------------------------------------------
__global__ void mask_topcap(const float* __restrict__ xsum,
                            float* __restrict__ xrec) {
  __shared__ float red[256];
  __shared__ float norms[10];
  __shared__ int   active;
  const int b = blockIdx.x;
  const int t = threadIdx.x;
  const float* xb = xsum + (size_t)b * PARFLAT;

  for (int c = 0; c < 10; ++c) {
    const float v = xb[c * 256 + t];
    red[t] = v * v;
    __syncthreads();
    for (int s = 128; s > 0; s >>= 1) {
      if (t < s) red[t] += red[t + s];
      __syncthreads();
    }
    if (t == 0) norms[c] = red[0];
    __syncthreads();
  }
  if (t == 0) {
    float best = norms[0]; int bi = 0;
    for (int c = 1; c < 10; ++c)
      if (norms[c] > best) { best = norms[c]; bi = c; }  // first-max on ties
    active = bi;
  }
  __syncthreads();
  for (int c = 0; c < 10; ++c)
    xrec[(size_t)b * PARFLAT + c * 256 + t] = (c == active) ? xb[c * 256 + t] : 0.f;
}

// ---------------------------------------------------------------------------
// WMMA GEMM: C[M,N] = act( A[M,K] @ W[K,N] + bias[N] ).  fp32 in memory,
// bf16 in the matrix pipes with f32 accumulation (16x16x32 is 8x the
// FLOP/instruction of the f32 16x16x4 path; the GEMM is compute-bound).
//
// Block: 128 threads = 4 wave32s; tile 64(M) x 128(N), K-step 32.
// Each wave computes a 32x64 sub-tile = 2 A-frags x 4 B-frags = 8 WMMAs/step.
// LDS tiles are *bf16* (converted once at staging), laid out so every
// fragment loads as two 16B ds_load_b128:
//   As  [row][k]   row-major      (A frag: K chunks {kbA..+7} and {kbA+16..+23})
//   BsT [col][k]   K-major/column (B frag: K chunks {kbB..+7} and {kbB+16..+23})
// Rows padded to 40 elements (80 B) -> all rows 16B-aligned, banks spread.
// ACT: 0 = relu, 1 = sigmoid.
// ---------------------------------------------------------------------------
struct Frag { union { bf16x16 v; uint4 q[2]; }; };

template<int ACT>
__global__ void gemm_bias_act(const float* __restrict__ A,
                              const float* __restrict__ W,
                              const float* __restrict__ bias,
                              float* __restrict__ C,
                              int M, int K, int N) {
  __shared__ __align__(16) __bf16 As [64][40];
  __shared__ __align__(16) __bf16 BsT[128][40];

  const int nBase = blockIdx.x * 128;
  const int mBase = blockIdx.y * 64;   // M = 512, always a multiple of 64
  const int tid   = threadIdx.x;
  const int lane  = tid & 31;
  const int wave  = tid >> 5;
  const int wm    = (wave >> 1) * 32;  // wave sub-tile origin in block tile
  const int wn    = (wave & 1)  * 64;
  const int l16   = lane & 15;
  const int half  = lane >> 4;         // lane-half selects K interleave
  const int kbA   = half * 8;          // A frag: K in {kbA..+7} U {kbA+16..+23}
  const int kbB   = half * 8;          // B frag: K in {kbB..+7} U {kbB+16..+23}

  f32x8 acc[2][4] = {};

  const int kTiles = (K + 31) / 32;
  for (int kt = 0; kt < kTiles; ++kt) {
    const int k0 = kt * 32;

    // prefetch next weight tile region into GL2 (global_prefetch_b8)
    if (kt + 1 < kTiles && nBase + tid < N)
      __builtin_prefetch(W + (size_t)(k0 + 32) * N + nBase + tid, 0, 1);

    // ---- stage A tile (64 x 32) as bf16, float4-coalesced along K --------
    for (int idx = tid; idx < 64 * 8; idx += 128) {      // 8 quads per row
      const int r  = idx >> 3;
      const int c4 = (idx & 7) * 4;
      __bf16 h[4];
      #pragma unroll
      for (int u = 0; u < 4; ++u) {
        const int gk = k0 + c4 + u;
        h[u] = (__bf16)((gk < K) ? A[(size_t)(mBase + r) * K + gk] : 0.f);
      }
      *(uint2*)&As[r][c4] = *(const uint2*)h;
    }
    // ---- stage B tile (32 x 128) transposed to BsT[col][k], bf16 ---------
    for (int idx = tid; idx < 128 * 8; idx += 128) {     // lanes -> columns
      const int c  = idx & 127;
      const int r4 = (idx >> 7) * 4;
      const int gc = nBase + c;
      __bf16 h[4];
      #pragma unroll
      for (int u = 0; u < 4; ++u) {
        const int gk = k0 + r4 + u;
        h[u] = (__bf16)((gk < K && gc < N) ? W[(size_t)gk * N + gc] : 0.f);
      }
      *(uint2*)&BsT[c][r4] = *(const uint2*)h;
    }
    __syncthreads();

    // ---- fragments: two ds_load_b128 each, zero conversions --------------
    Frag fa[2], fb[4];
    #pragma unroll
    for (int mi = 0; mi < 2; ++mi) {
      const int row = wm + mi * 16 + l16;
      fa[mi].q[0] = *(const uint4*)&As[row][kbA];
      fa[mi].q[1] = *(const uint4*)&As[row][kbA + 16];
    }
    #pragma unroll
    for (int ni = 0; ni < 4; ++ni) {
      const int col = wn + ni * 16 + l16;
      fb[ni].q[0] = *(const uint4*)&BsT[col][kbB];
      fb[ni].q[1] = *(const uint4*)&BsT[col][kbB + 16];
    }
    #pragma unroll
    for (int mi = 0; mi < 2; ++mi)
      #pragma unroll
      for (int ni = 0; ni < 4; ++ni)
        acc[mi][ni] = __builtin_amdgcn_wmma_f32_16x16x32_bf16(
            false, fa[mi].v, false, fb[ni].v,
            (short)0, acc[mi][ni], false, false);
    __syncthreads();
  }

  // ---- epilogue: bias + activation, bounds-checked stores ----------------
  #pragma unroll
  for (int mi = 0; mi < 2; ++mi) {
    #pragma unroll
    for (int ni = 0; ni < 4; ++ni) {
      const int col = nBase + wn + ni * 16 + l16;
      if (col < N) {
        const float bv = bias[col];
        const int rbase = mBase + wm + mi * 16 + half * 8; // D: VGPR r -> M=r / M=8+r
        #pragma unroll
        for (int r = 0; r < 8; ++r) {
          float v = acc[mi][ni][r] + bv;
          v = (ACT == 0) ? fmaxf(v, 0.f) : 1.f / (1.f + __expf(-v));
          C[(size_t)(rbase + r) * N + col] = v;
        }
      }
    }
  }
}

// ---------------------------------------------------------------------------
extern "C" void kernel_launch(void* const* d_in, const int* in_sizes, int n_in,
                              void* d_out, int out_size, void* d_ws, size_t ws_size,
                              hipStream_t stream) {
  (void)in_sizes; (void)n_in; (void)out_size; (void)ws_size;

  const float* x     = (const float*)d_in[0];
  const float* C1    = (const float*)d_in[1];
  const float* W1    = (const float*)d_in[2];
  const float* b1    = (const float*)d_in[3];
  const float* Cskip = (const float*)d_in[4];
  const float* C2    = (const float*)d_in[5];
  const float* W2    = (const float*)d_in[6];
  const float* b2    = (const float*)d_in[7];
  const float* Wr1   = (const float*)d_in[8];
  const float* br1   = (const float*)d_in[9];
  const float* Wr2   = (const float*)d_in[10];
  const float* br2   = (const float*)d_in[11];
  const float* Wr3   = (const float*)d_in[12];
  const float* br3   = (const float*)d_in[13];

  float* out  = (float*)d_out;                        // (512, 3072)
  float* xsum = out + (size_t)BATCH * INFLAT;         // (512, 2560) — 2nd output

  float* ws    = (float*)d_ws;
  float* xprim = ws;                                  // 512*3072
  float* xrec  = xprim + (size_t)BATCH * INFLAT;      // 512*2560
  float* h1    = xrec  + (size_t)BATCH * PARFLAT;     // 512*5000
  float* h2    = h1    + (size_t)BATCH * 5000;        // 512*4000

  // primary capsule layer (relu)
  caps_layer<12, false><<<dim3(BATCH, 48), 64, 0, stream>>>(x, C1, W1, b1, nullptr, xprim);
  // parent capsule layer + fused skip  ->  x_sum (directly into d_out)
  caps_layer<10, true ><<<dim3(BATCH, 40), 64, 0, stream>>>(xprim, C2, W2, b2, Cskip, xsum);
  // top-1 capsule masking
  mask_topcap<<<BATCH, 256, 0, stream>>>(xsum, xrec);
  // reconstruction MLP: 2560 -> 5000 -> 4000 -> 3072 (bf16 WMMA, f32 accum)
  gemm_bias_act<0><<<dim3((5000 + 127) / 128, BATCH / 64), 128, 0, stream>>>(xrec, Wr1, br1, h1, BATCH, 2560, 5000);
  gemm_bias_act<0><<<dim3((4000 + 127) / 128, BATCH / 64), 128, 0, stream>>>(h1,   Wr2, br2, h2, BATCH, 5000, 4000);
  gemm_bias_act<1><<<dim3((3072 + 127) / 128, BATCH / 64), 128, 0, stream>>>(h2,   Wr3, br3, out, BATCH, 4000, 3072);
}
